// GlobalMHSAWithPosBias_46420006535749
// MI455X (gfx1250) — matrix operand order repaired and checked by reading, not compiled, not run on hardware
//
#include <hip/hip_runtime.h>
#include <hip/hip_bf16.h>
#include <math.h>

typedef __attribute__((ext_vector_type(16))) __bf16 v16bf;
typedef __attribute__((ext_vector_type(8)))  __bf16 v8bf;
typedef __attribute__((ext_vector_type(8)))  float  v8f;

#define B_     128
#define C_     512
#define NHW    196
#define NP     224      // padded N: 7*32, 14 tiles of 16
#define NHEAD  8
#define HD     64
#define OC3    1536

// ---------- WMMA helpers ----------

__device__ __forceinline__ v8f zero_v8f() {
    v8f z;
#pragma unroll
    for (int i = 0; i < 8; i++) z[i] = 0.0f;
    return z;
}

__device__ __forceinline__ v8f wmma_bf16(v16bf a, v16bf b, v8f c) {
    // D = A(16x32 bf16) x B(32x16 bf16) + C(f32)
    return __builtin_amdgcn_wmma_f32_16x16x32_bf16(
        false, a, false, b, (short)0, c, false, false);
}

// A-matrix (16x32 bf16), source row-major [m][k], ld in elements.
// lanes 0-15: M=lane, K = {k0..k0+7, k0+16..k0+23}
// lanes 16-31: M=lane-16, K = {k0+8..k0+15, k0+24..k0+31}
__device__ __forceinline__ v16bf load_mat_a(const __bf16* __restrict__ p0, int ld) {
    int lane = threadIdx.x & 31;
    int hl = lane >> 4;
    int m  = lane & 15;
    const __bf16* p = p0 + m * ld + hl * 8;
    union { v16bf v; v8bf h[2]; } u;
    u.h[0] = *(const v8bf*)(p);
    u.h[1] = *(const v8bf*)(p + 16);
    return u.v;
}

// B-matrix (32x16 bf16), source stored as [n][k] row-major (K contiguous), ld in elements.
// lanes 0-15: N=lane, K = {k0..k0+7, k0+16..k0+23} ; lanes 16-31: N=lane-16, K = {k0+8..k0+15, k0+24..k0+31}
__device__ __forceinline__ v16bf load_mat_b(const __bf16* __restrict__ p0, int ld) {
    int lane = threadIdx.x & 31;
    int hl = lane >> 4;
    int n  = lane & 15;
    const __bf16* p = p0 + n * ld + hl * 8;
    union { v16bf v; v8bf h[2]; } u;
    u.h[0] = *(const v8bf*)(p);
    u.h[1] = *(const v8bf*)(p + 16);
    return u.v;
}

// ---------- Prep kernels ----------

// xT[b][n][c] = bf16(x[b][c][n]) (zero-padded rows n >= 196)
__global__ void prep_xT(const float* __restrict__ x, __bf16* __restrict__ xT) {
    long i = (long)blockIdx.x * blockDim.x + threadIdx.x;
    if (i >= (long)B_ * NP * C_) return;
    int c = (int)(i % C_);
    long t = i / C_;
    int n = (int)(t % NP);
    int b = (int)(t / NP);
    float v = (n < NHW) ? x[((long)b * C_ + c) * NHW + n] : 0.0f;
    xT[i] = (__bf16)v;
}

// w'[o][c] = bf16(w[o][c] * gamma[o] * rsqrt(var[o]+eps))
__global__ void prep_w(const float* __restrict__ w, const float* __restrict__ gamma,
                       const float* __restrict__ var, __bf16* __restrict__ wout,
                       int O, int Cc) {
    long i = (long)blockIdx.x * blockDim.x + threadIdx.x;
    if (i >= (long)O * Cc) return;
    int o = (int)(i / Cc);
    float inv = gamma[o] * rsqrtf(var[o] + 1e-5f);
    wout[i] = (__bf16)(w[i] * inv);
}

__global__ void prep_bvec(const float* __restrict__ gamma, const float* __restrict__ beta,
                          const float* __restrict__ mean, const float* __restrict__ var,
                          float* __restrict__ bout, int O) {
    int o = blockIdx.x * blockDim.x + threadIdx.x;
    if (o < O) {
        float inv = gamma[o] * rsqrtf(var[o] + 1e-5f);
        bout[o] = beta[o] - mean[o] * inv;
    }
}

// bm[h][n][m] = attention_biases[h][idxs[n][m]] (n,m < 196) else -1e30 (kills padded cols)
__global__ void prep_bias_mat(const float* __restrict__ ab, const int* __restrict__ idxs,
                              float* __restrict__ bmout, int nrel) {
    long i = (long)blockIdx.x * blockDim.x + threadIdx.x;
    if (i >= (long)NHEAD * NP * NP) return;
    int m = (int)(i % NP);
    long t = i / NP;
    int n = (int)(t % NP);
    int h = (int)(t / NP);
    float v = -1e30f;
    if (n < NHW && m < NHW) v = ab[h * nrel + idxs[n * NHW + m]];
    bmout[i] = v;
}

// ---------- Kernel 1: fused QKV GEMM (+BN1), scatter to q/k/v layouts ----------
// grid (14 n-tiles, 3 o-groups, 128 batch), 8 waves/block, each wave: 16n x 64o tile
__global__ __launch_bounds__(256) void qkv_gemm(
    const __bf16* __restrict__ xT, const __bf16* __restrict__ w1,
    const float* __restrict__ b1,
    __bf16* __restrict__ qb, __bf16* __restrict__ kb, __bf16* __restrict__ vb) {
    int ntile = blockIdx.x;
    int b     = blockIdx.z;
    int wave  = threadIdx.x >> 5;
    int lane  = threadIdx.x & 31;
    int obase = (blockIdx.y * 8 + wave) * 64;

    const __bf16* Abase = xT + ((long)b * NP + ntile * 16) * C_;

    v8f acc[4];
#pragma unroll
    for (int t = 0; t < 4; t++) acc[t] = zero_v8f();

    for (int k0 = 0; k0 < C_; k0 += 32) {
        v16bf a = load_mat_a(Abase + k0, C_);
#pragma unroll
        for (int t = 0; t < 4; t++) {
            v16bf bm = load_mat_b(w1 + (long)(obase + t * 16) * C_ + k0, C_);
            acc[t] = wmma_bf16(a, bm, acc[t]);
        }
    }

    int hl  = lane >> 4;
    int col = lane & 15;
#pragma unroll
    for (int t = 0; t < 4; t++) {
        int o  = obase + t * 16 + col;
        int s  = o / C_;          // 0=q 1=k 2=v
        int oc = o % C_;
        int h  = oc / HD;
        int d  = oc % HD;
        float bias = b1[o];
        long bh = (long)b * NHEAD + h;
#pragma unroll
        for (int r = 0; r < 8; r++) {
            int row = ntile * 16 + r + hl * 8;    // n index (may be padded, buffer sized NP)
            __bf16 bv = (__bf16)(acc[t][r] + bias);
            if (s == 0)      qb[(bh * NP + row) * HD + d] = bv;
            else if (s == 1) kb[(bh * NP + row) * HD + d] = bv;
            else             vb[(bh * HD + d) * NP + row] = bv;
        }
    }
}

// ---------- Kernel 2: attention (scores + bias + softmax + P*v^T + GELU) ----------
// grid (13 n-tiles, 8 heads, 128 batch), one wave per block
__global__ __launch_bounds__(32) void attn_kernel(
    const __bf16* __restrict__ qb, const __bf16* __restrict__ kb,
    const __bf16* __restrict__ vb, const float* __restrict__ bmat,
    __bf16* __restrict__ oT) {
    __shared__ __bf16 Pb[16 * NP];

    int ntile = blockIdx.x;
    int h     = blockIdx.y;
    int b     = blockIdx.z;
    int lane  = threadIdx.x & 31;
    int hl    = lane >> 4;
    int l15   = lane & 15;
    long bh   = (long)b * NHEAD + h;

    const __bf16* Aq = qb + (bh * NP + ntile * 16) * HD;

    // S[16 x 224] strip = q^T k
    v8f s[14];
#pragma unroll
    for (int t = 0; t < 14; t++) s[t] = zero_v8f();

#pragma unroll
    for (int ks = 0; ks < 2; ks++) {
        v16bf a = load_mat_a(Aq + ks * 32, HD);
#pragma unroll
        for (int t = 0; t < 14; t++) {
            v16bf bm = load_mat_b(kb + (bh * NP + t * 16) * HD + ks * 32, HD);
            s[t] = wmma_bf16(a, bm, s[t]);
        }
    }

    const float scale = 0.125f;  // hd^-0.5
#pragma unroll
    for (int t = 0; t < 14; t++) {
#pragma unroll
        for (int r = 0; r < 8; r++) {
            int row = ntile * 16 + r + hl * 8;
            int m   = t * 16 + l15;
            s[t][r] = s[t][r] * scale + bmat[((long)h * NP + row) * NP + m];
        }
    }

    // softmax over m (row data lives in one 16-lane half, 14 tiles x 8 row-slots)
#pragma unroll
    for (int r = 0; r < 8; r++) {
        float mx = -1e30f;
#pragma unroll
        for (int t = 0; t < 14; t++) mx = fmaxf(mx, s[t][r]);
#pragma unroll
        for (int off = 8; off >= 1; off >>= 1) mx = fmaxf(mx, __shfl_xor(mx, off, 32));
        float sum = 0.0f;
#pragma unroll
        for (int t = 0; t < 14; t++) {
            float e = __expf(s[t][r] - mx);
            s[t][r] = e;
            sum += e;
        }
#pragma unroll
        for (int off = 8; off >= 1; off >>= 1) sum += __shfl_xor(sum, off, 32);
        float rinv = 1.0f / sum;
#pragma unroll
        for (int t = 0; t < 14; t++) s[t][r] *= rinv;
    }

    // P -> LDS (row-major 16 x 224 bf16)
#pragma unroll
    for (int t = 0; t < 14; t++) {
#pragma unroll
        for (int r = 0; r < 8; r++) {
            int row = r + hl * 8;
            Pb[row * NP + t * 16 + l15] = (__bf16)s[t][r];
        }
    }
    __syncthreads();

    // O = P * v^T : 16n x 64d, K over m = 224
    v8f oacc[4];
#pragma unroll
    for (int t = 0; t < 4; t++) oacc[t] = zero_v8f();

#pragma unroll
    for (int k0 = 0; k0 < NP; k0 += 32) {
        union { v16bf v; v8bf hh[2]; } ua;
        int base = l15 * NP + k0 + hl * 8;
        ua.hh[0] = *(const v8bf*)&Pb[base];
        ua.hh[1] = *(const v8bf*)&Pb[base + 16];
#pragma unroll
        for (int t = 0; t < 4; t++) {
            v16bf bm = load_mat_b(vb + (bh * HD + t * 16) * NP + k0, NP);
            oacc[t] = wmma_bf16(ua.v, bm, oacc[t]);
        }
    }

    // GELU(exact) + scatter through LeViT reshape (h,n,d) -> (c, n')
#pragma unroll
    for (int t = 0; t < 4; t++) {
        int d = t * 16 + l15;
#pragma unroll
        for (int r = 0; r < 8; r++) {
            int n = ntile * 16 + r + hl * 8;
            if (n < NHW) {
                float vv = oacc[t][r];
                float g = 0.5f * vv * (1.0f + erff(vv * 0.70710678118f));
                int flat = h * (NHW * HD) + n * HD + d;
                int c  = flat / NHW;
                int np = flat % NHW;
                oT[((long)b * NP + np) * C_ + c] = (__bf16)g;
            }
        }
    }
}

// ---------- Kernel 3: projection GEMM (+BN2) ----------
// grid (13 n-tiles, 128 batch), 8 waves/block, wave = 16n x 64o tile
__global__ __launch_bounds__(256) void proj_gemm(
    const __bf16* __restrict__ oT, const __bf16* __restrict__ w2,
    const float* __restrict__ b2, float* __restrict__ out) {
    int ntile = blockIdx.x;
    int b     = blockIdx.y;
    int wave  = threadIdx.x >> 5;
    int lane  = threadIdx.x & 31;
    int obase = wave * 64;

    const __bf16* Abase = oT + ((long)b * NP + ntile * 16) * C_;

    v8f acc[4];
#pragma unroll
    for (int t = 0; t < 4; t++) acc[t] = zero_v8f();

    for (int k0 = 0; k0 < C_; k0 += 32) {
        v16bf a = load_mat_a(Abase + k0, C_);
#pragma unroll
        for (int t = 0; t < 4; t++) {
            v16bf bm = load_mat_b(w2 + (long)(obase + t * 16) * C_ + k0, C_);
            acc[t] = wmma_bf16(a, bm, acc[t]);
        }
    }

    int hl  = lane >> 4;
    int col = lane & 15;
#pragma unroll
    for (int t = 0; t < 4; t++) {
        int o = obase + t * 16 + col;
        float bias = b2[o];
#pragma unroll
        for (int r = 0; r < 8; r++) {
            int n = ntile * 16 + r + hl * 8;
            if (n < NHW) out[((long)b * C_ + o) * NHW + n] = acc[t][r] + bias;
        }
    }
}

// ---------- Launch ----------

extern "C" void kernel_launch(void* const* d_in, const int* in_sizes, int n_in,
                              void* d_out, int out_size, void* d_ws, size_t ws_size,
                              hipStream_t stream) {
    const float* x      = (const float*)d_in[0];
    const float* qkv_w  = (const float*)d_in[1];
    const float* g1     = (const float*)d_in[2];
    const float* be1    = (const float*)d_in[3];
    const float* mu1    = (const float*)d_in[4];
    const float* va1    = (const float*)d_in[5];
    const float* proj_w = (const float*)d_in[6];
    const float* g2     = (const float*)d_in[7];
    const float* be2    = (const float*)d_in[8];
    const float* mu2    = (const float*)d_in[9];
    const float* va2    = (const float*)d_in[10];
    const float* ab     = (const float*)d_in[11];
    const int*   idxs   = (const int*)d_in[12];
    float* out = (float*)d_out;
    int nrel = in_sizes[11] / NHEAD;

    // workspace layout (oT aliases xT: xT is dead after qkv_gemm)
    char* w = (char*)d_ws;
    size_t off = 0;
    auto take = [&](size_t bytes) {
        size_t o = off;
        off += (bytes + 255) & ~(size_t)255;
        return o;
    };
    size_t xT_off = take((size_t)B_ * NP * C_ * 2);         // also oT
    size_t qb_off = take((size_t)B_ * NHEAD * NP * HD * 2);
    size_t kb_off = take((size_t)B_ * NHEAD * NP * HD * 2);
    size_t vb_off = take((size_t)B_ * NHEAD * HD * NP * 2);
    size_t w1_off = take((size_t)OC3 * C_ * 2);
    size_t b1_off = take((size_t)OC3 * 4);
    size_t w2_off = take((size_t)C_ * C_ * 2);
    size_t b2_off = take((size_t)C_ * 4);
    size_t bm_off = take((size_t)NHEAD * NP * NP * 4);

    __bf16* xT = (__bf16*)(w + xT_off);
    __bf16* oT = (__bf16*)(w + xT_off);
    __bf16* qb = (__bf16*)(w + qb_off);
    __bf16* kb = (__bf16*)(w + kb_off);
    __bf16* vb = (__bf16*)(w + vb_off);
    __bf16* w1 = (__bf16*)(w + w1_off);
    float*  b1 = (float*)(w + b1_off);
    __bf16* w2 = (__bf16*)(w + w2_off);
    float*  b2 = (float*)(w + b2_off);
    float*  bm = (float*)(w + bm_off);

    // prep
    {
        long n = (long)B_ * NP * C_;
        prep_xT<<<dim3((unsigned)((n + 255) / 256)), 256, 0, stream>>>(x, xT);
    }
    {
        long n = (long)OC3 * C_;
        prep_w<<<dim3((unsigned)((n + 255) / 256)), 256, 0, stream>>>(qkv_w, g1, va1, w1, OC3, C_);
    }
    {
        long n = (long)C_ * C_;
        prep_w<<<dim3((unsigned)((n + 255) / 256)), 256, 0, stream>>>(proj_w, g2, va2, w2, C_, C_);
    }
    prep_bvec<<<dim3((OC3 + 255) / 256), 256, 0, stream>>>(g1, be1, mu1, va1, b1, OC3);
    prep_bvec<<<dim3((C_ + 255) / 256), 256, 0, stream>>>(g2, be2, mu2, va2, b2, C_);
    {
        long n = (long)NHEAD * NP * NP;
        prep_bias_mat<<<dim3((unsigned)((n + 255) / 256)), 256, 0, stream>>>(ab, idxs, bm, nrel);
    }

    // GEMM 1: qkv (+BN1) -> q, k, v
    qkv_gemm<<<dim3(NP / 16, OC3 / 512, B_), 256, 0, stream>>>(xT, w1, b1, qb, kb, vb);

    // attention + GELU -> oT (aliases xT)
    attn_kernel<<<dim3(13, NHEAD, B_), 32, 0, stream>>>(qb, kb, vb, bm, oT);

    // GEMM 2: proj (+BN2) -> out
    proj_gemm<<<dim3(13, B_, 1), 256, 0, stream>>>(oT, w2, b2, out);
}
